// SingleHeadAttention_1786706395425
// MI455X (gfx1250) — compile-verified, matrix operand order checked by reading, not run on hardware
//
#include <hip/hip_runtime.h>
#include <hip/hip_bf16.h>
#include <math.h>

// Problem constants
#define NB 8
#define NL 2048
#define ND 512

typedef __bf16 bf16;
typedef __attribute__((ext_vector_type(16))) __bf16 v16bf;
typedef __attribute__((ext_vector_type(8)))  __bf16 v8bf;
typedef __attribute__((ext_vector_type(8)))  float  v8f;

// f32 -> bf16, round-to-nearest-even (used only OUTSIDE GEMM inner loops)
__device__ __forceinline__ bf16 f2bf(float f) {
    unsigned u = __builtin_bit_cast(unsigned, f);
    unsigned r = u + 0x7FFFu + ((u >> 16) & 1u);
    unsigned short h = (unsigned short)(r >> 16);
    return __builtin_bit_cast(bf16, h);
}

// A/B fragment for v_wmma_f32_16x16x32_bf16 (wave32).
// Per ISA 7.12.2: lane holds row(A)/col(B) = lane%16; with half = lane>>4,
// element e in [0,8) -> k = 8*half + e ; e in [8,16) -> k = 16 + 8*half + (e-8).
// `p` must already point at  base_row + 8*half  (k0 folded in by caller).
__device__ __forceinline__ v16bf frag_from_bf16(const bf16* __restrict__ p) {
    v8bf lo = *(const v8bf*)(p);        // k = 8h .. 8h+7
    v8bf hi = *(const v8bf*)(p + 16);   // k = 16+8h .. 16+8h+7
    return __builtin_shufflevector(lo, hi, 0,1,2,3,4,5,6,7,8,9,10,11,12,13,14,15);
}

__device__ __forceinline__ v8f wmma_bf16(v16bf a, v16bf b, v8f c) {
    return __builtin_amdgcn_wmma_f32_16x16x32_bf16(false, a, false, b, (short)0, c,
                                                   false, false);
}

// C/D layout (f32 16x16): lane = n + 16*(m/8), vgpr r = m%8.
__device__ __forceinline__ void st_bf(bf16* __restrict__ C, int ldc, int mb, int n,
                                      const v8f& a) {
#pragma unroll
    for (int r = 0; r < 8; ++r) C[(size_t)(mb + r) * ldc + n] = f2bf(a[r]);
}
__device__ __forceinline__ void st_bf_t(bf16* __restrict__ C, int ldm, int mb, int n,
                                        const v8f& a) {
#pragma unroll
    for (int r = 0; r < 8; ++r) C[(size_t)n * ldm + (mb + r)] = f2bf(a[r]);
}

// ---------------------------------------------------------------------------
// Kernel 0: elementwise f32 -> bf16 (bandwidth-bound, one pass)
// ---------------------------------------------------------------------------
__global__ __launch_bounds__(256) void cvt_f32_bf16(const float* __restrict__ src,
                                                    bf16* __restrict__ dst, int n) {
    int i = (blockIdx.x * 256 + threadIdx.x) * 8;
    if (i + 8 <= n) {
#pragma unroll
        for (int j = 0; j < 8; ++j) dst[i + j] = f2bf(src[i + j]);
    }
}

// ---------------------------------------------------------------------------
// Kernel 1: projection  C[b,m,n] = sum_k X[b,m,k] * W[n,k]   (x @ W.T)
// All-bf16 operands. Each wave computes a 32x32 tile (4 WMMA / 4 loads).
// WG = 8 waves (2x4) covers 64x128. Grid (M/64, N/128, B). M=NL,N=ND,K=ND.
// ---------------------------------------------------------------------------
__global__ __launch_bounds__(256) void proj_kernel(
    const bf16* __restrict__ X, const bf16* __restrict__ W,
    bf16* __restrict__ Out, int transpose_out)
{
    const int M = NL, N = ND, K = ND;
    int lane = threadIdx.x & 31, wave = threadIdx.x >> 5;
    int l15 = lane & 15, half = lane >> 4;
    int m0 = (blockIdx.x * 2 + (wave >> 2)) * 32;
    int n0 = (blockIdx.y * 4 + (wave & 3)) * 32;
    int b  = blockIdx.z;

    const bf16* a0p = X + ((size_t)b * M + m0 + l15) * K + half * 8;
    const bf16* a1p = a0p + (size_t)16 * K;
    const bf16* b0p = W + (size_t)(n0 + l15) * K + half * 8;
    const bf16* b1p = b0p + (size_t)16 * K;

    v8f c00 = {}, c01 = {}, c10 = {}, c11 = {};
#pragma unroll 4
    for (int k = 0; k < K; k += 32) {
        v16bf a0 = frag_from_bf16(a0p + k);
        v16bf a1 = frag_from_bf16(a1p + k);
        v16bf b0 = frag_from_bf16(b0p + k);
        v16bf b1 = frag_from_bf16(b1p + k);
        c00 = wmma_bf16(a0, b0, c00);
        c01 = wmma_bf16(a0, b1, c01);
        c10 = wmma_bf16(a1, b0, c10);
        c11 = wmma_bf16(a1, b1, c11);
    }
    bf16* C = Out + (size_t)b * M * N;
    int mb = m0 + half * 8, n = n0 + l15;
    if (!transpose_out) {
        st_bf(C, N, mb,      n,      c00);
        st_bf(C, N, mb,      n + 16, c01);
        st_bf(C, N, mb + 16, n,      c10);
        st_bf(C, N, mb + 16, n + 16, c11);
    } else { // [N][M]
        st_bf_t(C, M, mb,      n,      c00);
        st_bf_t(C, M, mb,      n + 16, c01);
        st_bf_t(C, M, mb + 16, n,      c10);
        st_bf_t(C, M, mb + 16, n + 16, c11);
    }
}

// ---------------------------------------------------------------------------
// Kernel 2: scores S[b,m,n] = (sum_k qh[b,m,k]*kh[b,n,k]) / sqrt(D), masked.
// Writes f32 into the attn region of d_out. Grid (L/64, L/128, B).
// ---------------------------------------------------------------------------
__device__ __forceinline__ void st_scores(float* __restrict__ C, int mb, int n,
                                          int len, const v8f& a) {
    const float scale = 0.04419417382415922f; // 1/sqrt(512)
    bool ncol = (n < len);
#pragma unroll
    for (int r = 0; r < 8; ++r) {
        int m = mb + r;
        C[(size_t)m * NL + n] =
            (ncol && (m < len)) ? a[r] * scale : -__builtin_inff();
    }
}

__global__ __launch_bounds__(256) void scores_kernel(
    const bf16* __restrict__ QH, const bf16* __restrict__ KH,
    const int* __restrict__ lens, float* __restrict__ S)
{
    const int M = NL, N = NL, K = ND;
    int lane = threadIdx.x & 31, wave = threadIdx.x >> 5;
    int l15 = lane & 15, half = lane >> 4;
    int m0 = (blockIdx.x * 2 + (wave >> 2)) * 32;
    int n0 = (blockIdx.y * 4 + (wave & 3)) * 32;
    int b  = blockIdx.z;
    int len = lens[b];

    const bf16* a0p = QH + ((size_t)b * M + m0 + l15) * K + half * 8;
    const bf16* a1p = a0p + (size_t)16 * K;
    const bf16* b0p = KH + ((size_t)b * N + n0 + l15) * K + half * 8;
    const bf16* b1p = b0p + (size_t)16 * K;

    v8f c00 = {}, c01 = {}, c10 = {}, c11 = {};
#pragma unroll 4
    for (int k = 0; k < K; k += 32) {
        v16bf a0 = frag_from_bf16(a0p + k);
        v16bf a1 = frag_from_bf16(a1p + k);
        v16bf b0 = frag_from_bf16(b0p + k);
        v16bf b1 = frag_from_bf16(b1p + k);
        c00 = wmma_bf16(a0, b0, c00);
        c01 = wmma_bf16(a0, b1, c01);
        c10 = wmma_bf16(a1, b0, c10);
        c11 = wmma_bf16(a1, b1, c11);
    }
    float* C = S + (size_t)b * NL * NL;
    int mb = m0 + half * 8, n = n0 + l15;
    st_scores(C, mb,      n,      len, c00);
    st_scores(C, mb,      n + 16, len, c01);
    st_scores(C, mb + 16, n,      len, c10);
    st_scores(C, mb + 16, n + 16, len, c11);
}

// ---------------------------------------------------------------------------
// Kernel 3: row softmax in place on S[b,row,:], plus bf16 copy for attn@V.
// Fully-masked rows -> 0 (matches reference NaN->0). One 256-thread block/row.
// ---------------------------------------------------------------------------
__global__ __launch_bounds__(256) void softmax_kernel(float* __restrict__ S,
                                                      bf16* __restrict__ Sbf)
{
    __shared__ float red[256];
    int t = threadIdx.x;
    size_t rowoff = ((size_t)blockIdx.y * NL + blockIdx.x) * NL;
    float* p  = S   + rowoff;
    bf16*  pb = Sbf + rowoff;

    float x[8];
    float mx = -__builtin_inff();
#pragma unroll
    for (int i = 0; i < 8; ++i) { x[i] = p[t + 256 * i]; mx = fmaxf(mx, x[i]); }

    red[t] = mx; __syncthreads();
#pragma unroll
    for (int s = 128; s > 0; s >>= 1) {
        if (t < s) red[t] = fmaxf(red[t], red[t + s]);
        __syncthreads();
    }
    float rmax = red[0];
    __syncthreads();

    if (rmax == -__builtin_inff()) {       // all-masked row -> zeros
#pragma unroll
        for (int i = 0; i < 8; ++i) {
            p[t + 256 * i]  = 0.0f;
            pb[t + 256 * i] = f2bf(0.0f);
        }
        return;
    }

    float sum = 0.0f;
#pragma unroll
    for (int i = 0; i < 8; ++i) { x[i] = __expf(x[i] - rmax); sum += x[i]; }

    red[t] = sum; __syncthreads();
#pragma unroll
    for (int s = 128; s > 0; s >>= 1) {
        if (t < s) red[t] += red[t + s];
        __syncthreads();
    }
    float inv = 1.0f / red[0];
#pragma unroll
    for (int i = 0; i < 8; ++i) {
        float v = x[i] * inv;
        p[t + 256 * i]  = v;
        pb[t + 256 * i] = f2bf(v);
    }
}

// ---------------------------------------------------------------------------
// Kernel 4: O[b,m,n] = sum_k attn_bf[b,m,k] * vhT[b,n,k]  (vhT: [B,D,L] bf16)
// Grid (L/64, D/128, B). M=NL, N=ND, K=NL.
// ---------------------------------------------------------------------------
__global__ __launch_bounds__(256) void av_kernel(
    const bf16* __restrict__ Abf, const bf16* __restrict__ VT,
    bf16* __restrict__ O)
{
    const int M = NL, N = ND, K = NL;
    int lane = threadIdx.x & 31, wave = threadIdx.x >> 5;
    int l15 = lane & 15, half = lane >> 4;
    int m0 = (blockIdx.x * 2 + (wave >> 2)) * 32;
    int n0 = (blockIdx.y * 4 + (wave & 3)) * 32;
    int b  = blockIdx.z;

    const bf16* a0p = Abf + ((size_t)b * M + m0 + l15) * K + half * 8;
    const bf16* a1p = a0p + (size_t)16 * K;
    const bf16* b0p = VT + ((size_t)b * N + n0 + l15) * K + half * 8;
    const bf16* b1p = b0p + (size_t)16 * K;

    v8f c00 = {}, c01 = {}, c10 = {}, c11 = {};
#pragma unroll 2
    for (int k = 0; k < K; k += 32) {
        v16bf a0 = frag_from_bf16(a0p + k);
        v16bf a1 = frag_from_bf16(a1p + k);
        v16bf b0 = frag_from_bf16(b0p + k);
        v16bf b1 = frag_from_bf16(b1p + k);
        c00 = wmma_bf16(a0, b0, c00);
        c01 = wmma_bf16(a0, b1, c01);
        c10 = wmma_bf16(a1, b0, c10);
        c11 = wmma_bf16(a1, b1, c11);
    }
    bf16* C = O + (size_t)b * M * N;
    int mb = m0 + half * 8, n = n0 + l15;
    st_bf(C, N, mb,      n,      c00);
    st_bf(C, N, mb,      n + 16, c01);
    st_bf(C, N, mb + 16, n,      c10);
    st_bf(C, N, mb + 16, n + 16, c11);
}

// ---------------------------------------------------------------------------
// Kernel 5: fc (WMMA into LDS) + residual + LayerNorm, fused per 16-row stripe.
// AO bf16 [B,L,D]; Wb bf16 [D,D]; Q residual f32; out f32 [B,L,D].
// 8 waves: each wave owns 4 of 32 column tiles, reusing one A-frag per k step.
// Grid (L/16, B).
// ---------------------------------------------------------------------------
__global__ __launch_bounds__(256) void fc_ln_kernel(
    const bf16* __restrict__ AO, const bf16* __restrict__ Wb,
    const float* __restrict__ Q, const float* __restrict__ gamma,
    const float* __restrict__ beta, float* __restrict__ Out)
{
    __shared__ float smem[16][ND]; // 32 KB
    int lane = threadIdx.x & 31, wave = threadIdx.x >> 5;
    int l15 = lane & 15, half = lane >> 4;
    int b  = blockIdx.y;
    int m0 = blockIdx.x * 16;

    const bf16* ap = AO + ((size_t)b * NL + m0 + l15) * ND + half * 8;
    const bf16* bp0 = Wb + (size_t)((wave * 4 + 0) * 16 + l15) * ND + half * 8;
    const bf16* bp1 = bp0 + (size_t)16 * ND;
    const bf16* bp2 = bp1 + (size_t)16 * ND;
    const bf16* bp3 = bp2 + (size_t)16 * ND;

    v8f c0 = {}, c1 = {}, c2 = {}, c3 = {};
#pragma unroll 4
    for (int k = 0; k < ND; k += 32) {
        v16bf a = frag_from_bf16(ap + k);
        c0 = wmma_bf16(a, frag_from_bf16(bp0 + k), c0);
        c1 = wmma_bf16(a, frag_from_bf16(bp1 + k), c1);
        c2 = wmma_bf16(a, frag_from_bf16(bp2 + k), c2);
        c3 = wmma_bf16(a, frag_from_bf16(bp3 + k), c3);
    }
    int mb = half * 8;
#pragma unroll
    for (int r = 0; r < 8; ++r) {
        smem[mb + r][(wave * 4 + 0) * 16 + l15] = c0[r];
        smem[mb + r][(wave * 4 + 1) * 16 + l15] = c1[r];
        smem[mb + r][(wave * 4 + 2) * 16 + l15] = c2[r];
        smem[mb + r][(wave * 4 + 3) * 16 + l15] = c3[r];
    }
    __syncthreads();

    const float inv_d = 1.0f / (float)ND;
#pragma unroll
    for (int rr = 0; rr < 2; ++rr) {
        int row = wave * 2 + rr;
        const float* qrow = Q + ((size_t)b * NL + m0 + row) * ND;
        float xv[16], s = 0.0f, ss = 0.0f;
#pragma unroll
        for (int i = 0; i < 16; ++i) {
            int c = lane + 32 * i;
            float x = smem[row][c] + qrow[c];
            xv[i] = x; s += x; ss += x * x;
        }
#pragma unroll
        for (int off = 16; off > 0; off >>= 1) {
            s  += __shfl_xor(s, off, 32);
            ss += __shfl_xor(ss, off, 32);
        }
        float mu   = s * inv_d;
        float var  = ss * inv_d - mu * mu;
        float rstd = rsqrtf(var + 1e-6f);
        float* orow = Out + ((size_t)b * NL + m0 + row) * ND;
#pragma unroll
        for (int i = 0; i < 16; ++i) {
            int c = lane + 32 * i;
            orow[c] = (xv[i] - mu) * rstd * gamma[c] + beta[c];
        }
    }
}

// ---------------------------------------------------------------------------
extern "C" void kernel_launch(void* const* d_in, const int* in_sizes, int n_in,
                              void* d_out, int out_size, void* d_ws, size_t ws_size,
                              hipStream_t stream) {
    (void)in_sizes; (void)n_in; (void)out_size; (void)ws_size;

    const float* q    = (const float*)d_in[0];
    const float* kk   = (const float*)d_in[1];
    const float* vv   = (const float*)d_in[2];
    const int*   lens = (const int*)d_in[3];
    const float* Wq   = (const float*)d_in[4];
    const float* Wk   = (const float*)d_in[5];
    const float* Wv   = (const float*)d_in[6];
    const float* Wfc  = (const float*)d_in[7];
    const float* gam  = (const float*)d_in[8];
    const float* bet  = (const float*)d_in[9];

    float* out  = (float*)d_out;                        // [B,L,D]
    float* attn = out + (size_t)NB * NL * ND;           // [B,L,L]

    const size_t BLD  = (size_t)NB * NL * ND * sizeof(bf16); // 16 MiB
    const size_t WDD  = (size_t)ND * ND * sizeof(bf16);      // 0.5 MiB
    const int    nBLD = NB * NL * ND;                        // 8,388,608
    const int    nWDD = ND * ND;                             // 262,144

    char* ws = (char*)d_ws;
    // Region 0 (4*BLD == B*L*L*2 bytes): qb,kb,qh,kh -- dead after scores;
    // exactly reused as attn_bf by softmax/av.
    bf16* qb      = (bf16*)(ws + 0 * BLD);
    bf16* kb      = (bf16*)(ws + 1 * BLD);
    bf16* qh      = (bf16*)(ws + 2 * BLD);
    bf16* kh      = (bf16*)(ws + 3 * BLD);
    bf16* attn_bf = (bf16*)(ws + 0 * BLD);  // aliases qb..kh (dead by then)
    // Region 1: vb, dead after V projection; reused as ao.
    bf16* vb = (bf16*)(ws + 4 * BLD);
    bf16* ao = (bf16*)(ws + 4 * BLD);       // aliases vb (dead by then)
    // Region 2: vhT [B,D,L]
    bf16* vt = (bf16*)(ws + 5 * BLD);
    // Region 3: weights
    bf16* wqb  = (bf16*)(ws + 6 * BLD + 0 * WDD);
    bf16* wkb  = (bf16*)(ws + 6 * BLD + 1 * WDD);
    bf16* wvb  = (bf16*)(ws + 6 * BLD + 2 * WDD);
    bf16* wfcb = (bf16*)(ws + 6 * BLD + 3 * WDD);

    dim3 blk(256);

    // One-shot conversions (bandwidth-bound, trivial cost)
    cvt_f32_bf16<<<dim3(nBLD / 2048), blk, 0, stream>>>(q,   qb,   nBLD);
    cvt_f32_bf16<<<dim3(nBLD / 2048), blk, 0, stream>>>(kk,  kb,   nBLD);
    cvt_f32_bf16<<<dim3(nBLD / 2048), blk, 0, stream>>>(vv,  vb,   nBLD);
    cvt_f32_bf16<<<dim3(nWDD / 2048), blk, 0, stream>>>(Wq,  wqb,  nWDD);
    cvt_f32_bf16<<<dim3(nWDD / 2048), blk, 0, stream>>>(Wk,  wkb,  nWDD);
    cvt_f32_bf16<<<dim3(nWDD / 2048), blk, 0, stream>>>(Wv,  wvb,  nWDD);
    cvt_f32_bf16<<<dim3(nWDD / 2048), blk, 0, stream>>>(Wfc, wfcb, nWDD);

    // Projections (pure bf16 WMMA)
    proj_kernel<<<dim3(NL / 64, ND / 128, NB), blk, 0, stream>>>(qb, wqb, qh, 0);
    proj_kernel<<<dim3(NL / 64, ND / 128, NB), blk, 0, stream>>>(kb, wkb, kh, 0);
    proj_kernel<<<dim3(NL / 64, ND / 128, NB), blk, 0, stream>>>(vb, wvb, vt, 1);

    // Scores + mask -> attn region of d_out (f32)
    scores_kernel<<<dim3(NL / 64, NL / 128, NB), blk, 0, stream>>>(qh, kh, lens, attn);

    // Softmax in place + bf16 copy (reuses region 0)
    softmax_kernel<<<dim3(NL, NB), blk, 0, stream>>>(attn, attn_bf);

    // attn @ V (pure bf16 WMMA), ao reuses region 1
    av_kernel<<<dim3(NL / 64, ND / 128, NB), blk, 0, stream>>>(attn_bf, vt, ao);

    // fc + residual + LayerNorm
    fc_ln_kernel<<<dim3(NL / 16, NB), blk, 0, stream>>>(ao, wfcb, q, gam, bet, out);
}
